// ResRnn_24524263260685
// MI455X (gfx1250) — compile-verified, matrix operand order checked by reading, not run on hardware
//
#include <hip/hip_runtime.h>
#include <hip/hip_bf16.h>
#include <stdint.h>

#define SEQ   512
#define BATCH 64
#define IN_W  64
#define SW    1024
#define SWP   1032        // padded LDS row stride (ushorts): +16B to break bank conflicts
#define NBLK  64
#define TPB   128
#define COLS  16
#define LIN_F 0.99999f

typedef __attribute__((ext_vector_type(16))) __bf16 v16bf;
typedef __attribute__((ext_vector_type(8)))  float  v8f;
typedef __attribute__((ext_vector_type(4)))  int    v4i;
typedef __attribute__((address_space(3)))    v4i    lds_v4i;

#if defined(__has_builtin)
#if __has_builtin(__builtin_amdgcn_global_load_async_to_lds_b128)
#define RNN_HAS_ASYNC 1
#endif
#endif

__device__ __forceinline__ unsigned short f2bf(float f) {
  unsigned u = __float_as_uint(f);
  unsigned r = u + 0x7FFFu + ((u >> 16) & 1u);
  return (unsigned short)(r >> 16);
}

__device__ __forceinline__ unsigned pack2(float a, float b) {
#if __has_builtin(__builtin_amdgcn_cvt_pk_bf16_f32)
  typedef __attribute__((ext_vector_type(2))) __bf16 v2bf;
  union { v2bf v; unsigned u; } r;
  r.v = __builtin_amdgcn_cvt_pk_bf16_f32(a, b);
  return r.u;
#else
  return (unsigned)f2bf(a) | ((unsigned)f2bf(b) << 16);
#endif
}

__device__ __forceinline__ void wait_async0() {
#ifdef RNN_HAS_ASYNC
#if __has_builtin(__builtin_amdgcn_s_wait_asynccnt)
  __builtin_amdgcn_s_wait_asynccnt(0);
#else
  asm volatile("s_wait_asynccnt 0x0" ::: "memory");
#endif
#endif
}

// Device-scope grid barrier; counter monotonically increases, target advances by NBLK.
__device__ __forceinline__ void grid_sync(unsigned* bar, unsigned target) {
  __syncthreads();
  if (threadIdx.x == 0) {
    __threadfence();
    __hip_atomic_fetch_add(bar, 1u, __ATOMIC_RELEASE, __HIP_MEMORY_SCOPE_AGENT);
    while (__hip_atomic_load(bar, __ATOMIC_ACQUIRE, __HIP_MEMORY_SCOPE_AGENT) < target)
      __builtin_amdgcn_s_sleep(2);
  }
  __syncthreads();
  // Discard stale near-cache lines so we observe other WGPs' stores.
  asm volatile("global_inv scope:SCOPE_DEV\n\t"
               "s_wait_loadcnt 0x0" ::: "memory");
}

// Build a 16-bit WMMA A/B fragment per the CDNA5 VGPR layout:
// lanes 0-15: row=lane, K in {0..7, 16..23}; lanes 16-31: same row, K +8.
// The 16 bf16 values per lane are exactly two contiguous, 16B-aligned chunks.
__device__ __forceinline__ v16bf load_frag(const unsigned short* row, int kb, int hlf) {
  union { v16bf v; uint4 q[2]; } f;
  const unsigned short* p = row + kb + hlf * 8;
  f.q[0] = *(const uint4*)(p);        // K = kb + hlf*8 + 0..7      -> ds_load_b128
  f.q[1] = *(const uint4*)(p + 16);   // K = kb + hlf*8 + 16..23    -> ds_load_b128
  return f.v;
}

__global__ void resrnn_init(const float* __restrict__ init_stream,
                            float* __restrict__ S0, unsigned short* __restrict__ SB0,
                            unsigned* __restrict__ bar) {
  int i = blockIdx.x * blockDim.x + threadIdx.x;
  if (i == 0) *bar = 0u;
  if (i < BATCH * SW) {
    float v = (1.0f - LIN_F) * init_stream[i & (SW - 1)];
    S0[i]  = v;
    SB0[i] = f2bf(v);
  }
}

__global__ void __launch_bounds__(TPB)
resrnn_kernel(const float* __restrict__ inp,
              const float* __restrict__ W1, const float* __restrict__ b1,
              const float* __restrict__ W2, const float* __restrict__ b2,
              float* __restrict__ outp,
              float* __restrict__ S0, float* __restrict__ S1,
              unsigned short* __restrict__ SB0, unsigned short* __restrict__ SB1,
              unsigned short* __restrict__ Hg,
              unsigned* __restrict__ bar)
{
  extern __shared__ unsigned short smem[];
  unsigned short* aAct = smem;                 // BATCH x SWP
  unsigned short* w1l  = smem + BATCH * SWP;   // COLS  x SWP
  unsigned short* w2l  = w1l + COLS * SWP;     // COLS  x SWP

  const int tid  = threadIdx.x;
  const int wave = tid >> 5;
  const int lane = tid & 31;
  const int m    = lane & 15;
  const int hlf  = lane >> 4;
  const int rowBase = wave * 16;
  const int colBase = blockIdx.x * COLS;

  // ---- one-time: stage this WG's bf16 weight slices (rows o = colBase..+15) into LDS
  for (int idx = tid; idx < COLS * (SW / 4); idx += TPB) {
    int r = idx >> 8;             // SW/4 == 256
    int k = (idx & 255) << 2;
    float4 w1v = *(const float4*)(W1 + (size_t)(colBase + r) * SW + k);
    float4 w2v = *(const float4*)(W2 + (size_t)(colBase + r) * SW + k);
    *(uint2*)(w1l + r * SWP + k) = make_uint2(pack2(w1v.x, w1v.y), pack2(w1v.z, w1v.w));
    *(uint2*)(w2l + r * SWP + k) = make_uint2(pack2(w2v.x, w2v.y), pack2(w2v.z, w2v.w));
  }
  const float bias1 = b1[colBase + m];
  const float bias2 = b2[colBase + m];
  const float OM = 1.0f - LIN_F;
  __syncthreads();

  unsigned target = NBLK;

  for (int t = 0; t < SEQ; ++t) {
    const float* inp_t = inp + (size_t)t * BATCH * IN_W;
    float*          Sprev  = (t & 1) ? S1  : S0;
    float*          Snext  = (t & 1) ? S0  : S1;
    unsigned short* SBprev = (t & 1) ? SB1 : SB0;
    unsigned short* SBnext = (t & 1) ? SB0 : SB1;

    if (t + 1 < SEQ)
      __builtin_prefetch(inp + (size_t)(t + 1) * BATCH * IN_W + tid * 16, 0, 0);

    // ---- stage shifted = [input_t | SBprev[:, :960]] as bf16 into LDS
    // (a) input columns: f32 -> bf16 convert
    for (int idx = tid; idx < BATCH * (IN_W / 4); idx += TPB) {   // 1024 float4 chunks
      int b = idx >> 4;
      int k = (idx & 15) << 2;
      float4 v = *(const float4*)(inp_t + b * IN_W + k);
      *(uint2*)(aAct + b * SWP + k) = make_uint2(pack2(v.x, v.y), pack2(v.z, v.w));
    }
    // (b) recurrent columns: straight bf16 copy (async global->LDS when available)
    for (int idx = tid; idx < BATCH * 128; idx += TPB) {
      int b = idx >> 7;
      int c = idx & 127;
      if (c < 120) {                          // 960 ushorts = 120 x b128 per row
        int k = IN_W + (c << 3);
#ifdef RNN_HAS_ASYNC
        __builtin_amdgcn_global_load_async_to_lds_b128(
            (v4i*)(SBprev + b * SW + (k - IN_W)),
            (lds_v4i*)(uint32_t)(uintptr_t)(aAct + b * SWP + k), 0, 0);
#else
        *(uint4*)(aAct + b * SWP + k) = *(const uint4*)(SBprev + b * SW + (k - IN_W));
#endif
      }
    }
    wait_async0();
    __syncthreads();

    // ---- GEMM1 (software-pipelined): h-tile = relu(shifted @ W1.T + b1)
    v8f acc;
#pragma unroll
    for (int j = 0; j < 8; ++j) acc[j] = 0.0f;
    {
      const unsigned short* arow = aAct + (rowBase + m) * SWP;
      const unsigned short* brow = w1l + m * SWP;
      v16bf af = load_frag(arow, 0, hlf);
      v16bf bf = load_frag(brow, 0, hlf);
#pragma unroll 2
      for (int kb = 0; kb < SW - 32; kb += 32) {
        v16bf af2 = load_frag(arow, kb + 32, hlf);
        v16bf bf2 = load_frag(brow, kb + 32, hlf);
        acc = __builtin_amdgcn_wmma_f32_16x16x32_bf16(false, af, false, bf,
                                                      (short)0, acc, false, false);
        af = af2; bf = bf2;
      }
      acc = __builtin_amdgcn_wmma_f32_16x16x32_bf16(false, af, false, bf,
                                                    (short)0, acc, false, false);
    }
#pragma unroll
    for (int r = 0; r < 8; ++r) {
      int b = rowBase + r + hlf * 8;                 // C/D layout: M = r (+8 for lanes 16-31)
      float hv = fmaxf(acc[r] + bias1, 0.0f);
      Hg[b * SW + colBase + m] = f2bf(hv);
    }

    grid_sync(bar, target); target += NBLK;

    // ---- stage full hidden H (bf16) into LDS (async global->LDS when available)
    for (int idx = tid; idx < BATCH * (SW / 8); idx += TPB) {
      int b  = idx >> 7;                            // 128 b128 chunks per row
      int kq = (idx & 127) << 3;
#ifdef RNN_HAS_ASYNC
      __builtin_amdgcn_global_load_async_to_lds_b128(
          (v4i*)(Hg + b * SW + kq),
          (lds_v4i*)(uint32_t)(uintptr_t)(aAct + b * SWP + kq), 0, 0);
#else
      *(uint4*)(aAct + b * SWP + kq) = *(const uint4*)(Hg + b * SW + kq);
#endif
    }
    wait_async0();
    __syncthreads();

    // ---- GEMM2 (software-pipelined): y-tile = h @ W2.T + b2; new = LIN*shifted + (1-LIN)*y
    v8f acc2;
#pragma unroll
    for (int j = 0; j < 8; ++j) acc2[j] = 0.0f;
    {
      const unsigned short* arow = aAct + (rowBase + m) * SWP;
      const unsigned short* brow = w2l + m * SWP;
      v16bf af = load_frag(arow, 0, hlf);
      v16bf bf = load_frag(brow, 0, hlf);
#pragma unroll 2
      for (int kb = 0; kb < SW - 32; kb += 32) {
        v16bf af2 = load_frag(arow, kb + 32, hlf);
        v16bf bf2 = load_frag(brow, kb + 32, hlf);
        acc2 = __builtin_amdgcn_wmma_f32_16x16x32_bf16(false, af, false, bf,
                                                       (short)0, acc2, false, false);
        af = af2; bf = bf2;
      }
      acc2 = __builtin_amdgcn_wmma_f32_16x16x32_bf16(false, af, false, bf,
                                                     (short)0, acc2, false, false);
    }
    float* out_t = outp + (size_t)t * BATCH * SW;
#pragma unroll
    for (int r = 0; r < 8; ++r) {
      int b   = rowBase + r + hlf * 8;
      int col = colBase + m;
      float sh = (col < IN_W) ? inp_t[b * IN_W + col] : Sprev[b * SW + (col - IN_W)];
      float nv = LIN_F * sh + OM * (acc2[r] + bias2);
      Snext[b * SW + col]  = nv;          // f32 stream (residual precision)
      SBnext[b * SW + col] = f2bf(nv);    // bf16 shadow (next step's matmul operand)
      out_t[b * SW + col]  = nv;
    }

    grid_sync(bar, target); target += NBLK;
  }
}

extern "C" void kernel_launch(void* const* d_in, const int* in_sizes, int n_in,
                              void* d_out, int out_size, void* d_ws, size_t ws_size,
                              hipStream_t stream) {
  (void)in_sizes; (void)n_in; (void)out_size; (void)ws_size;
  const float* inp  = (const float*)d_in[0];
  const float* strm = (const float*)d_in[1];
  const float* W1   = (const float*)d_in[2];
  const float* b1   = (const float*)d_in[3];
  const float* W2   = (const float*)d_in[4];
  const float* b2   = (const float*)d_in[5];
  float* outp = (float*)d_out;

  char* ws = (char*)d_ws;
  float*          S0  = (float*)(ws);                       // 256 KB
  float*          S1  = (float*)(ws + 256 * 1024);          // 256 KB
  unsigned short* Hg  = (unsigned short*)(ws + 512 * 1024); // 128 KB
  unsigned short* SB0 = (unsigned short*)(ws + 640 * 1024); // 128 KB
  unsigned short* SB1 = (unsigned short*)(ws + 768 * 1024); // 128 KB
  unsigned*       bar = (unsigned*)(ws + 896 * 1024);

  resrnn_init<<<256, 256, 0, stream>>>(strm, S0, SB0, bar);

  size_t lds_bytes = (size_t)(BATCH + 2 * COLS) * SWP * sizeof(unsigned short); // ~194 KB
  resrnn_kernel<<<NBLK, TPB, lds_bytes, stream>>>(inp, W1, b1, W2, b2, outp,
                                                  S0, S1, SB0, SB1, Hg, bar);
}
